// Decoder_7876970020957
// MI455X (gfx1250) — matrix operand order repaired and checked by reading, not compiled, not run on hardware
//
#include <hip/hip_runtime.h>
#include <hip/hip_bf16.h>
#include <math.h>

// ---------------- problem constants (match reference) ----------------
#define VV 32000
#define EE 256
#define HH 512
#define BB 32
#define SS 512
#define TT 128
#define SOS_IDX 1

typedef __attribute__((ext_vector_type(16))) _Float16 v16h;
typedef __attribute__((ext_vector_type(8)))  _Float16 v8h;
typedef __attribute__((ext_vector_type(8)))  float    v8f;

// ---------------------------------------------------------------------
// CDNA5 async global->LDS copy (ASYNCcnt-tracked DMA, ISA §15.18 op 98).
// VDST VGPR carries the per-lane LDS byte offset; VADDR the 64-bit global
// address. Generic pointers to __shared__ carry the LDS offset in bits[31:0].
// ---------------------------------------------------------------------
__device__ __forceinline__ void async_copy16(void* lds_ptr, const void* gptr) {
    unsigned lds_off = (unsigned)(size_t)lds_ptr;
    asm volatile("global_load_async_to_lds_b128 %0, %1, off"
                 :: "v"(lds_off), "v"(gptr)
                 : "memory");
}
__device__ __forceinline__ void wait_asynccnt0() {
    asm volatile("s_wait_asynccnt 0x0" ::: "memory");
}

// ---------------------------------------------------------------------
// f32 -> f16 conversion (grid-stride)
// ---------------------------------------------------------------------
__global__ void cvt_f32_f16(const float* __restrict__ in, _Float16* __restrict__ out, size_t n) {
    size_t i = (size_t)blockIdx.x * blockDim.x + threadIdx.x;
    size_t stride = (size_t)gridDim.x * blockDim.x;
    for (; i < n; i += stride) out[i] = (_Float16)in[i];
}

// ---------------------------------------------------------------------
// gi0[j] = b_ih_l0[j] + dot(W_ih_l0[j, :E], embedding[SOS])   (j < 3H)
// constant across batch and time (decoder input is always SOS embedding)
// ---------------------------------------------------------------------
__global__ void gi0_kernel(const float* __restrict__ emb,
                           const float* __restrict__ Wih,
                           const float* __restrict__ bih,
                           float* __restrict__ gi0) {
    int j = blockIdx.x * blockDim.x + threadIdx.x;
    if (j >= 3 * HH) return;
    const float* e = emb + (size_t)SOS_IDX * EE;
    const float* w = Wih + (size_t)j * EE;
    float s = bih[j];
    for (int k = 0; k < EE; ++k) s += w[k] * e[k];
    gi0[j] = s;
}

// ---------------------------------------------------------------------
// h_state[l,b,n] = bp[n] + Wp[n,:] . concat(enc_h_n[2l+0,b,:], enc_h_n[2l+1,b,:])
// ---------------------------------------------------------------------
__global__ void init_h0(const float* __restrict__ ehn,
                        const float* __restrict__ Wp,
                        const float* __restrict__ bp,
                        float* __restrict__ h) {
    int idx = blockIdx.x * blockDim.x + threadIdx.x;   // over 2*B*H
    if (idx >= 2 * BB * HH) return;
    int n = idx & (HH - 1);
    int b = (idx >> 9) & (BB - 1);
    int l = idx >> 14;
    const float* w = Wp + (size_t)n * (2 * HH);
    const float* f = ehn + ((size_t)(2 * l + 0) * BB + b) * HH;
    const float* r = ehn + ((size_t)(2 * l + 1) * BB + b) * HH;
    float s = bp[n];
    for (int k = 0; k < HH; ++k) s += w[k] * f[k];
    for (int k = 0; k < HH; ++k) s += w[HH + k] * r[k];
    h[idx] = s;
}

// ---------------------------------------------------------------------
// WMMA GEMM with double-buffered async-LDS staging:
//   C[m,n] = sum_k A[m,k]*Bw[n,k] (+ bias[n])
//   A: M x K f16 row-major, Bw: N x K f16 row-major, C: f32.
//   Block = 256 threads = 8 waves; block tile 64(M) x 128(N); K-chunk 32.
//   Each K-chunk tile (A 4KB + B 8KB) is DMA'd global->LDS exactly once via
//   GLOBAL_LOAD_ASYNC_TO_LDS_B128 (next chunk overlaps current WMMAs), then
//   waves read fragments with 16B-aligned ds_load_b128 in the §7.12.2 layout.
//   Requires M%64==0, N%128==0, K%32==0 (true for all call sites).
//   If permLog2>0: logical row m = t*B + b (B=1<<permLog2) is stored at
//   row b*permT + t  -> performs the (T,B,V)->(B,T,V) transpose on store.
// ---------------------------------------------------------------------
__global__ __launch_bounds__(256)
void wmma_gemm_nt(const _Float16* __restrict__ A,
                  const _Float16* __restrict__ Bw,
                  const float* __restrict__ bias,
                  float* __restrict__ C,
                  int M, int N, int K, int permLog2, int permT) {
    __shared__ _Float16 As[2][64][32];    // [buf][m-local][k-local]  4KB each
    __shared__ _Float16 Bs[2][128][32];   // [buf][n-local][k-local]  8KB each

    const int tid  = threadIdx.x;
    const int lane = tid & 31;
    const int wave = tid >> 5;
    const int wr = wave >> 2;        // 0..1  (M direction)
    const int wc = wave & 3;         // 0..3  (N direction)
    const int m0b = blockIdx.x * 64;
    const int n0b = blockIdx.y * 128;
    const int lr = lane & 15;
    const int hi = lane >> 4;        // half-wave select

    // cooperative staging indices: thread covers 16B (8 halfs) per op
    const int srow = tid >> 2;        // 0..63
    const int sseg = (tid & 3) * 8;   // 0,8,16,24 (halfs)

    v8f acc[2][2] = {};

    // stage first K-chunk (async, ASYNCcnt-tracked)
    {
        async_copy16(&As[0][srow][sseg],      A  + (size_t)(m0b + srow) * K + sseg);
        async_copy16(&Bs[0][srow][sseg],      Bw + (size_t)(n0b + srow) * K + sseg);
        async_copy16(&Bs[0][srow + 64][sseg], Bw + (size_t)(n0b + srow + 64) * K + sseg);
    }

    int buf = 0;
    for (int kc = 0; kc < K; kc += 32, buf ^= 1) {
        wait_asynccnt0();            // this wave's staging of `buf` is complete
        __syncthreads();             // everyone's staging complete / reads of buf^1 done

        if (kc + 32 < K) {           // overlap next-chunk DMA with this chunk's WMMAs
            int kn = kc + 32;
            __builtin_prefetch(Bw + (size_t)(n0b + srow) * K + kn + 32, 0, 0);
            async_copy16(&As[buf ^ 1][srow][sseg],      A  + (size_t)(m0b + srow) * K + kn + sseg);
            async_copy16(&Bs[buf ^ 1][srow][sseg],      Bw + (size_t)(n0b + srow) * K + kn + sseg);
            async_copy16(&Bs[buf ^ 1][srow + 64][sseg], Bw + (size_t)(n0b + srow + 64) * K + kn + sseg);
        }

        // fragment loads from LDS (16-bit A/B layouts, 16B-aligned -> ds_load_b128)
        v16h a[2], b[2];
#pragma unroll
        for (int i = 0; i < 2; ++i) {
            const _Float16* ap = &As[buf][wr * 32 + i * 16 + lr][hi * 8];
            v8h lo  = *(const v8h*)(ap);
            v8h hi8 = *(const v8h*)(ap + 16);
#pragma unroll
            for (int e = 0; e < 8; ++e) { a[i][e] = lo[e]; a[i][e + 8] = hi8[e]; }
        }
#pragma unroll
        for (int j = 0; j < 2; ++j) {
            const _Float16* bp = &Bs[buf][wc * 32 + j * 16 + lr][hi * 8];
            v8h lo  = *(const v8h*)(bp);
            v8h hi8 = *(const v8h*)(bp + 16);
#pragma unroll
            for (int e = 0; e < 8; ++e) { b[j][e] = lo[e]; b[j][e + 8] = hi8[e]; }
        }
#pragma unroll
        for (int i = 0; i < 2; ++i)
#pragma unroll
            for (int j = 0; j < 2; ++j)
                acc[i][j] = __builtin_amdgcn_wmma_f32_16x16x32_f16(
                    false, a[i], false, b[j], (short)0, acc[i][j], false, false);
    }

    // Epilogue: C/D layout -> VGPR r: lanes 0-15 = row r, lanes 16-31 = row r+8
#pragma unroll
    for (int i = 0; i < 2; ++i) {
#pragma unroll
        for (int j = 0; j < 2; ++j) {
            int col = n0b + wc * 32 + j * 16 + lr;
            float bb = bias ? bias[col] : 0.0f;
            int rbase = m0b + wr * 32 + i * 16 + hi * 8;
#pragma unroll
            for (int r = 0; r < 8; ++r) {
                int mm = rbase + r;
                size_t orow = (size_t)mm;
                if (permLog2 > 0) {   // shift/mask permute: B is a power of two
                    int tt  = mm >> permLog2;
                    int bb2 = mm & ((1 << permLog2) - 1);
                    orow = (size_t)bb2 * permT + tt;
                }
                C[orow * (size_t)N + col] = acc[i][j][r] + bb;
            }
        }
    }
}

// ---------------------------------------------------------------------
// Wave-per-output GEMV: out[b,j] = sum_k X[b,k] * W[j,k]
//   (recurrent-path GEMMs; kept in f32 to preserve GRU state precision)
// ---------------------------------------------------------------------
__global__ __launch_bounds__(256)
void gemv_nt(const float* __restrict__ X, const float* __restrict__ W,
             float* __restrict__ out, int Bc, int N, int K) {
    int lane = threadIdx.x & 31;
    int gw = (int)((blockIdx.x * blockDim.x + threadIdx.x) >> 5);
    if (gw >= Bc * N) return;
    int b = gw / N;
    int j = gw - b * N;
    const float* x = X + (size_t)b * K;
    const float* w = W + (size_t)j * K;
    float s = 0.0f;
    for (int k = lane; k < K; k += 32) s += x[k] * w[k];
#pragma unroll
    for (int off = 16; off > 0; off >>= 1) s += __shfl_xor(s, off, 32);
    if (lane == 0) out[(size_t)b * N + j] = s;
}

// ---------------------------------------------------------------------
// GRU pointwise, layer 0: gi is batch-invariant (precomputed gi0, bias folded)
// ---------------------------------------------------------------------
__global__ void gru_combine0(const float* __restrict__ gi0,
                             const float* __restrict__ gh,
                             const float* __restrict__ bhh,
                             float* __restrict__ h) {
    int idx = blockIdx.x * blockDim.x + threadIdx.x;   // B*H
    if (idx >= BB * HH) return;
    int n = idx & (HH - 1);
    int b = idx >> 9;
    const float* g = gh + (size_t)b * (3 * HH);
    float ir = gi0[n], iz = gi0[HH + n], in_ = gi0[2 * HH + n];
    float hr = g[n] + bhh[n];
    float hz = g[HH + n] + bhh[HH + n];
    float hn = g[2 * HH + n] + bhh[2 * HH + n];
    float r = 1.0f / (1.0f + __expf(-(ir + hr)));
    float z = 1.0f / (1.0f + __expf(-(iz + hz)));
    float nn = tanhf(in_ + r * hn);
    h[idx] = (1.0f - z) * nn + z * h[idx];
}

// GRU pointwise, layer 1: gi per-batch (raw), add both biases here
__global__ void gru_combine1(const float* __restrict__ gi,
                             const float* __restrict__ gh,
                             const float* __restrict__ bih,
                             const float* __restrict__ bhh,
                             float* __restrict__ h) {
    int idx = blockIdx.x * blockDim.x + threadIdx.x;
    if (idx >= BB * HH) return;
    int n = idx & (HH - 1);
    int b = idx >> 9;
    const float* gI = gi + (size_t)b * (3 * HH);
    const float* gH = gh + (size_t)b * (3 * HH);
    float ir = gI[n] + bih[n];
    float iz = gI[HH + n] + bih[HH + n];
    float in_ = gI[2 * HH + n] + bih[2 * HH + n];
    float hr = gH[n] + bhh[n];
    float hz = gH[HH + n] + bhh[HH + n];
    float hn = gH[2 * HH + n] + bhh[2 * HH + n];
    float r = 1.0f / (1.0f + __expf(-(ir + hr)));
    float z = 1.0f / (1.0f + __expf(-(iz + hz)));
    float nn = tanhf(in_ + r * hn);
    h[idx] = (1.0f - z) * nn + z * h[idx];
}

// ---------------------------------------------------------------------
// Attention: one block per batch element. scores -> softmax (LDS) -> ctx
// ---------------------------------------------------------------------
__global__ __launch_bounds__(512)
void attention_kernel(const float* __restrict__ proj,   // (B,S,H)
                      const float* __restrict__ enc,    // (B,S,2H)
                      const float* __restrict__ dec,    // (B,H)
                      float* __restrict__ ctx) {        // (B,2H)
    __shared__ float w[SS];
    __shared__ float decs[HH];
    __shared__ float red[16];
    int b = blockIdx.x, tid = threadIdx.x;
    int lane = tid & 31, wv = tid >> 5;                 // 16 waves
    decs[tid] = dec[(size_t)b * HH + tid];
    __syncthreads();

    // scores: wave-per-row dot products (coalesced over k)
    for (int s = wv; s < SS; s += 16) {
        const float* p = proj + ((size_t)b * SS + s) * HH;
        float acc = 0.0f;
        for (int k = lane; k < HH; k += 32) acc += p[k] * decs[k];
#pragma unroll
        for (int off = 16; off > 0; off >>= 1) acc += __shfl_xor(acc, off, 32);
        if (lane == 0) w[s] = acc;
    }
    __syncthreads();

    // softmax over S=512
    float v = w[tid];
    float m = v;
#pragma unroll
    for (int off = 16; off > 0; off >>= 1) m = fmaxf(m, __shfl_xor(m, off, 32));
    if (lane == 0) red[wv] = m;
    __syncthreads();
    float mx = red[0];
#pragma unroll
    for (int i = 1; i < 16; ++i) mx = fmaxf(mx, red[i]);
    float e = __expf(v - mx);
    float sum = e;
#pragma unroll
    for (int off = 16; off > 0; off >>= 1) sum += __shfl_xor(sum, off, 32);
    __syncthreads();
    if (lane == 0) red[wv] = sum;
    __syncthreads();
    float tot = 0.0f;
#pragma unroll
    for (int i = 0; i < 16; ++i) tot += red[i];
    w[tid] = e / tot;
    __syncthreads();

    // ctx[d] = sum_s w[s] * enc[b,s,d], thread handles d=tid and d=tid+512
    const float* eb = enc + (size_t)b * SS * (2 * HH);
    float c0 = 0.0f, c1 = 0.0f;
    for (int s = 0; s < SS; ++s) {
        float ws = w[s];
        const float* row = eb + (size_t)s * (2 * HH);
        c0 += ws * row[tid];
        c1 += ws * row[HH + tid];
    }
    ctx[(size_t)b * (2 * HH) + tid] = c0;
    ctx[(size_t)b * (2 * HH) + HH + tid] = c1;
}

// ---------------------------------------------------------------------
// comb_h[t*B+b, n] = f16( tanh( bc[n] + Wc[n,:H].dec[b] + Wc[n,H:3H].ctx[b] ) )
// wave-per-output; buffered so the vocab GEMM runs once, off the critical path
// ---------------------------------------------------------------------
__global__ __launch_bounds__(256)
void comb_kernel(const float* __restrict__ dec, const float* __restrict__ ctx,
                 const float* __restrict__ Wc, const float* __restrict__ bc,
                 _Float16* __restrict__ comb_h, int t) {
    int lane = threadIdx.x & 31;
    int gw = (int)((blockIdx.x * blockDim.x + threadIdx.x) >> 5);
    if (gw >= BB * HH) return;
    int b = gw >> 9;
    int n = gw & (HH - 1);
    const float* w = Wc + (size_t)n * (3 * HH);
    const float* d = dec + (size_t)b * HH;
    const float* c = ctx + (size_t)b * (2 * HH);
    float s = 0.0f;
    for (int k = lane; k < HH; k += 32) s += w[k] * d[k];
    for (int k = lane; k < 2 * HH; k += 32) s += w[HH + k] * c[k];
#pragma unroll
    for (int off = 16; off > 0; off >>= 1) s += __shfl_xor(s, off, 32);
    if (lane == 0)
        comb_h[((size_t)t * BB + b) * HH + n] = (_Float16)tanhf(s + bc[n]);
}

// ---------------------------------------------------------------------
extern "C" void kernel_launch(void* const* d_in, const int* in_sizes, int n_in,
                              void* d_out, int out_size, void* d_ws, size_t ws_size,
                              hipStream_t stream) {
    const float* enc_outputs = (const float*)d_in[0];
    const float* enc_h_n     = (const float*)d_in[1];
    const float* embedding   = (const float*)d_in[2];
    const float* W_ih_l0 = (const float*)d_in[3];
    const float* W_hh_l0 = (const float*)d_in[4];
    const float* b_ih_l0 = (const float*)d_in[5];
    const float* b_hh_l0 = (const float*)d_in[6];
    const float* W_ih_l1 = (const float*)d_in[7];
    const float* W_hh_l1 = (const float*)d_in[8];
    const float* b_ih_l1 = (const float*)d_in[9];
    const float* b_hh_l1 = (const float*)d_in[10];
    const float* Wp = (const float*)d_in[11];
    const float* bp = (const float*)d_in[12];
    const float* Wa = (const float*)d_in[13];
    const float* ba = (const float*)d_in[14];
    const float* Wc = (const float*)d_in[15];
    const float* bc = (const float*)d_in[16];
    const float* Wv = (const float*)d_in[17];
    const float* bv = (const float*)d_in[18];
    float* out = (float*)d_out;

    // -------- workspace bump allocator --------
    char* ws = (char*)d_ws;
    size_t off = 0;
    auto alloc = [&](size_t bytes) -> void* {
        void* p = ws + off;
        off += (bytes + 255) & ~(size_t)255;
        return p;
    };
    _Float16* Wv_h  = (_Float16*)alloc((size_t)VV * HH * 2);          // 32.8 MB
    _Float16* enc_h = (_Float16*)alloc((size_t)BB * SS * 2 * HH * 2); // 33.6 MB
    _Float16* Wa_h  = (_Float16*)alloc((size_t)HH * 2 * HH * 2);      //  1.0 MB
    float* proj   = (float*)alloc((size_t)BB * SS * HH * 4);          // 33.6 MB
    float* gi0    = (float*)alloc((size_t)3 * HH * 4);
    float* h0     = (float*)alloc((size_t)BB * HH * 4);
    float* h1     = (float*)alloc((size_t)BB * HH * 4);
    float* g0     = (float*)alloc((size_t)BB * 3 * HH * 4);
    float* g1     = (float*)alloc((size_t)BB * 3 * HH * 4);
    float* g2     = (float*)alloc((size_t)BB * 3 * HH * 4);
    float* ctx    = (float*)alloc((size_t)BB * 2 * HH * 4);
    _Float16* comb_h = (_Float16*)alloc((size_t)TT * BB * HH * 2);    //  4.2 MB

    // -------- one-time precompute --------
    cvt_f32_f16<<<4096, 256, 0, stream>>>(Wv, Wv_h, (size_t)VV * HH);
    cvt_f32_f16<<<4096, 256, 0, stream>>>(enc_outputs, enc_h, (size_t)BB * SS * 2 * HH);
    cvt_f32_f16<<<2048, 256, 0, stream>>>(Wa, Wa_h, (size_t)HH * 2 * HH);
    gi0_kernel<<<(3 * HH + 255) / 256, 256, 0, stream>>>(embedding, W_ih_l0, b_ih_l0, gi0);
    init_h0<<<(2 * BB * HH + 255) / 256, 256, 0, stream>>>(enc_h_n, Wp, bp, h0);

    // proj_enc = enc_outputs @ Wa.T + ba   (M=16384, N=512, K=1024) via WMMA
    wmma_gemm_nt<<<dim3((BB * SS) / 64, HH / 128), 256, 0, stream>>>(
        enc_h, Wa_h, ba, proj, BB * SS, HH, 2 * HH, 0, 0);

    // -------- sequential decode: recurrence only (vocab GEMM deferred) --------
    const int gemvBlocks = (BB * 3 * HH) / 8;   // 8 waves per 256-thread block
    for (int t = 0; t < TT; ++t) {
        gemv_nt<<<gemvBlocks, 256, 0, stream>>>(h0, W_hh_l0, g0, BB, 3 * HH, HH);
        gru_combine0<<<(BB * HH) / 256, 256, 0, stream>>>(gi0, g0, b_hh_l0, h0);
        gemv_nt<<<gemvBlocks, 256, 0, stream>>>(h0, W_ih_l1, g1, BB, 3 * HH, HH);
        gemv_nt<<<gemvBlocks, 256, 0, stream>>>(h1, W_hh_l1, g2, BB, 3 * HH, HH);
        gru_combine1<<<(BB * HH) / 256, 256, 0, stream>>>(g1, g2, b_ih_l1, b_hh_l1, h1);
        attention_kernel<<<BB, 512, 0, stream>>>(proj, enc_outputs, h1, ctx);
        comb_kernel<<<(BB * HH) / 8, 256, 0, stream>>>(h1, ctx, Wc, bc, comb_h, t);
    }

    // -------- deferred vocab projection: one big WMMA GEMM --------
    // logits (T*B=4096, V=32000, K=512); epilogue permutes rows (T,B)->(B,T)
    wmma_gemm_nt<<<dim3((TT * BB) / 64, VV / 128), 256, 0, stream>>>(
        comb_h, Wv_h, bv, out, TT * BB, VV, HH, 5 /*log2(B)*/, TT);
}
